// SE3_GraphEncoder_51857435132001
// MI455X (gfx1250) — matrix operand order repaired and checked by reading, not compiled, not run on hardware
//
#include <hip/hip_runtime.h>

// =====================================================================
// SE3 GraphEncoder for MI455X (gfx1250, wave32, WMMA).
//
// Compute-bound (~490 GFLOP of [M,K]x[K,256] GEMMs vs a few GB of HBM
// traffic at 23.3 TB/s), so all matmuls run through
// v_wmma_f32_16x16x32_bf16 with f32 accumulation. bf16 chosen over f16
// for exponent range (softplus/LN outputs).
//
// GEMM staging uses CDNA5 async global->LDS DMA
// (global_load_async_to_lds_b128 + s_wait_asynccnt) with LDS double
// buffering so the next K-chunk streams in while WMMA consumes the
// current one. Falls back to sync staging if the builtin is absent.
//
// Assumes params flatten depth-first in dict insertion order (w before b,
// None biases skipped). Needs ~1.0 GB of d_ws scratch.
// =====================================================================

typedef float  v8f     __attribute__((ext_vector_type(8)));
typedef __bf16 bf16x8  __attribute__((ext_vector_type(8)));
typedef __bf16 bf16x16 __attribute__((ext_vector_type(16)));

#define FC 256
#define NNODES 4000
#define NEDGES 32000
#define NGRAPH 64

#if defined(__has_builtin)
#if __has_builtin(__builtin_amdgcn_global_load_async_to_lds_b128)
#define ASYNC_LDS 1
#endif
#endif

#if defined(ASYNC_LDS)
// Builtin signature (from hipcc diagnostic): param0 = int-vec16 addrspace(1)*,
// param1 = int-vec16 addrspace(3)*, then imm offset, imm cpol.
typedef int v4i_ls __attribute__((vector_size(16)));
typedef __attribute__((address_space(1))) v4i_ls* as1_v4i;
typedef __attribute__((address_space(3))) v4i_ls* as3_v4i;
#endif

__device__ __forceinline__ void async_cp16(const __bf16* g, __bf16* l) {
#if defined(ASYNC_LDS)
  __builtin_amdgcn_global_load_async_to_lds_b128(
      (as1_v4i)(void*)const_cast<__bf16*>(g), (as3_v4i)(void*)l, 0, 0);
#else
  (void)g; (void)l;
#endif
}

__device__ __forceinline__ void wait_async_le6() {
#if defined(ASYNC_LDS)
#if __has_builtin(__builtin_amdgcn_s_wait_asynccnt)
  __builtin_amdgcn_s_wait_asynccnt(6);
#else
  asm volatile("s_wait_asynccnt 0x6" ::: "memory");
#endif
#endif
}

__device__ __forceinline__ void wait_async_le0() {
#if defined(ASYNC_LDS)
#if __has_builtin(__builtin_amdgcn_s_wait_asynccnt)
  __builtin_amdgcn_s_wait_asynccnt(0);
#else
  asm volatile("s_wait_asynccnt 0x0" ::: "memory");
#endif
#endif
}

__device__ __forceinline__ __bf16 f2bf(float f) {
  unsigned u = __float_as_uint(f);
  u += 0x7FFFu + ((u >> 16) & 1u);          // round-to-nearest-even
  unsigned short h = (unsigned short)(u >> 16);
  return __builtin_bit_cast(__bf16, h);
}

__device__ __forceinline__ float softplusf(float x) {
  return (x > 20.f) ? x : log1pf(__expf(x));
}

template <int EPI>
__device__ __forceinline__ float epi_apply(float x) {
  if (EPI == 1) return x / (1.f + __expf(-x));   // SiLU
  if (EPI == 2) return softplusf(x);             // Softplus
  return x;
}

// ---------------------------------------------------------------------
// WMMA GEMM: out[M,256] = A[M,Kp](bf16) @ Wt[256,Kp]^T(bf16) + bias
// Workgroup: 256 threads (8 wave32), tile 128 rows x 256 cols.
// Async-DMA double-buffered K staging; fragments per CDNA5 ISA layouts.
// ---------------------------------------------------------------------
template <int EPI>
__global__ void __launch_bounds__(256)
k_gemm(const __bf16* __restrict__ A, const __bf16* __restrict__ Bt,
       const float* __restrict__ bias, float* __restrict__ outF,
       __bf16* __restrict__ outH, int M, int Kp) {
  __shared__ __bf16 As[2][128 * 40];   // 32 + 8 pad -> 80B rows (16B aligned)
  __shared__ __bf16 Bs[2][256 * 40];

  const int tid  = threadIdx.x;
  const int w    = tid >> 5;
  const int lane = tid & 31;
  const int m0   = blockIdx.x * 128;

  v8f zero = {0.f, 0.f, 0.f, 0.f, 0.f, 0.f, 0.f, 0.f};
  v8f acc[16];
#pragma unroll
  for (int n = 0; n < 16; ++n) acc[n] = zero;

  const int arow = tid >> 1;            // 0..127
  const int akk  = (tid & 1) << 4;      // 0 or 16
  int agr = m0 + arow;
  if (agr >= M) agr = M - 1;  // clamp (dup row): keeps async op count uniform;
                              // rows >= M are never stored in the epilogue.
  const __bf16* gA = A + (size_t)agr * Kp + akk;
  const __bf16* gB = Bt + (size_t)tid * Kp;

  const int nch = Kp >> 5;

  auto stage = [&](int c) {
    const int kb = c << 5;
    const int b  = c & 1;
    __bf16* la = &As[b][arow * 40 + akk];
    __bf16* lb = &Bs[b][tid * 40];
#if defined(ASYNC_LDS)
    async_cp16(gA + kb,      la);
    async_cp16(gA + kb + 8,  la + 8);
    async_cp16(gB + kb,      lb);
    async_cp16(gB + kb + 8,  lb + 8);
    async_cp16(gB + kb + 16, lb + 16);
    async_cp16(gB + kb + 24, lb + 24);
#else
    const uint4* ap = reinterpret_cast<const uint4*>(gA + kb);
    uint4 a0 = ap[0], a1 = ap[1];
    const uint4* bp = reinterpret_cast<const uint4*>(gB + kb);
    uint4 b0 = bp[0], b1 = bp[1], b2 = bp[2], b3 = bp[3];
    reinterpret_cast<uint4*>(la)[0]     = a0;
    reinterpret_cast<uint4*>(la + 8)[0] = a1;
    uint4* bq = reinterpret_cast<uint4*>(lb);
    bq[0] = b0; bq[1] = b1; bq[2] = b2; bq[3] = b3;
#endif
  };

  const int afr  = (w << 4) + (lane & 15);
  const int aoff = (lane < 16) ? 0 : 8;   // A lane K-offsets {0,16} / {8,24}
  const int boff = (lane < 16) ? 0 : 8;   // B lane K-offsets {0,16} / {8,24}

  stage(0);

  for (int c = 0; c < nch; ++c) {
    if (c + 1 < nch) {
      stage(c + 1);      // overlap next chunk DMA with this chunk's WMMAs
      wait_async_le6();  // wave issued 6 async ops/chunk -> chunk c complete
    } else {
      wait_async_le0();
    }
    __syncthreads();

    const __bf16* as = As[c & 1];
    const __bf16* bs = Bs[c & 1];

    bf16x8 alo = *reinterpret_cast<const bf16x8*>(&as[afr * 40 + aoff]);
    bf16x8 ahi = *reinterpret_cast<const bf16x8*>(&as[afr * 40 + aoff + 16]);
    bf16x16 afrag = __builtin_shufflevector(
        alo, ahi, 0, 1, 2, 3, 4, 5, 6, 7, 8, 9, 10, 11, 12, 13, 14, 15);

    // software-pipelined N-tile loop: load frag n+1 before WMMA n
    const __bf16* bb0 = &bs[(lane & 15) * 40 + boff];
    bf16x8 blo = *reinterpret_cast<const bf16x8*>(bb0);
    bf16x8 bhi = *reinterpret_cast<const bf16x8*>(bb0 + 16);
    bf16x16 bfrag = __builtin_shufflevector(
        blo, bhi, 0, 1, 2, 3, 4, 5, 6, 7, 8, 9, 10, 11, 12, 13, 14, 15);
#pragma unroll
    for (int n = 0; n < 16; ++n) {
      bf16x16 bnext = bfrag;
      if (n < 15) {
        const __bf16* bb = &bs[(((n + 1) << 4) + (lane & 15)) * 40 + boff];
        bf16x8 nlo = *reinterpret_cast<const bf16x8*>(bb);
        bf16x8 nhi = *reinterpret_cast<const bf16x8*>(bb + 16);
        bnext = __builtin_shufflevector(
            nlo, nhi, 0, 1, 2, 3, 4, 5, 6, 7, 8, 9, 10, 11, 12, 13, 14, 15);
      }
      acc[n] = __builtin_amdgcn_wmma_f32_16x16x32_bf16(
          false, afrag, false, bfrag, (short)0, acc[n], false, false);
      bfrag = bnext;
    }
    __syncthreads();
  }

  // ---- epilogue: C layout lane L, vgpr j -> row j + 8*(L>=16), col L&15 ----
  const int colbase = lane & 15;
  const int rbase   = m0 + (w << 4) + ((lane >> 4) << 3);
#pragma unroll
  for (int n = 0; n < 16; ++n) {
    const int col = (n << 4) + colbase;
    const float bv = bias ? bias[col] : 0.f;
#pragma unroll
    for (int j = 0; j < 8; ++j) {
      const int r = rbase + j;
      if (r < M) {
        float v = epi_apply<EPI>(acc[n][j] + bv);
        size_t o = (size_t)r * 256 + col;
        if (outF) outF[o] = v;
        if (outH) outH[o] = f2bf(v);
      }
    }
  }
}

// ---------------------------------------------------------------------
// Weight prep: W[K][256] f32 -> Wt[256][Kp] bf16 (transposed, K padded)
// ---------------------------------------------------------------------
__global__ void k_wprep(const float* __restrict__ w, __bf16* __restrict__ wt,
                        int K, int Kp) {
  int idx = blockIdx.x * 256 + threadIdx.x;  // total 256*Kp
  int n = idx / Kp, k = idx % Kp;
  if (n >= 256) return;
  wt[(size_t)n * Kp + k] = (k < K) ? f2bf(w[(size_t)k * 256 + n]) : f2bf(0.f);
}

__global__ void k_zero(float* p, int n) {
  int i = blockIdx.x * 256 + threadIdx.x;
  if (i < n) p[i] = 0.f;
}

__global__ void k_cvt(const float* __restrict__ x, __bf16* __restrict__ o, int n) {
  int i = blockIdx.x * 256 + threadIdx.x;
  if (i < n) o[i] = f2bf(x[i]);
}

// x[4000,92] f32 -> [4000,96] bf16 zero-padded
__global__ void k_pack_x(const float* __restrict__ x, __bf16* __restrict__ out,
                         int rows) {
  int i = blockIdx.x * 256 + threadIdx.x;
  if (i >= rows * 96) return;
  int r = i / 96, c = i % 96;
  out[i] = (c < 92) ? f2bf(x[r * 92 + c]) : f2bf(0.f);
}

// RBF expansion -> bf16 [rows,256].  mode 0: t=-0.75/|edge_attr[e]| (rows=E)
// mode 1: t=src[e*3+i], row=i*E+e (i-major, rows=3E)
__global__ void k_rbf(const float* __restrict__ src, __bf16* __restrict__ out,
                      int E, int mode, float vmin, float vmax) {
  int row = blockIdx.x, c = threadIdx.x;
  float t;
  if (mode == 0) {
    float a = src[row * 3], b = src[row * 3 + 1], d = src[row * 3 + 2];
    t = -0.75f / sqrtf(a * a + b * b + d * d);
  } else {
    int i2 = row / E, e = row % E;
    t = src[e * 3 + i2];
  }
  float span = vmax - vmin;
  float gamma = 255.f / span;
  float ctr = vmin + (float)c * span * (1.f / 255.f);
  float d0 = t - ctr;
  out[(size_t)row * 256 + c] = f2bf(__expf(-gamma * d0 * d0));
}

// [nei_len(256) | lemb[i](32)] -> bf16 [3E,288] (i-major rows)
__global__ void k_cat288(const __bf16* __restrict__ nl,
                         const float* __restrict__ lemb,
                         __bf16* __restrict__ out, int E) {
  int row = blockIdx.x, c = threadIdx.x;  // blockDim = 288
  int i2 = row / E;
  __bf16 v = (c < 256) ? nl[(size_t)row * 256 + c]
                       : f2bf(lemb[i2 * 32 + (c - 256)]);
  out[(size_t)row * 288 + c] = v;
}

// edge layer concat: [kx[e] | ky[row] | ea[row]] -> bf16 [3E,768]
__global__ void k_cat768_edge(const __bf16* __restrict__ kx,
                              const __bf16* __restrict__ ky,
                              const __bf16* __restrict__ ea,
                              __bf16* __restrict__ out, int E) {
  int row = blockIdx.x, c = threadIdx.x;  // blockDim = 768
  int e = row % E;
  __bf16 v;
  if (c < 256)      v = kx[(size_t)e * 256 + c];
  else if (c < 512) v = ky[(size_t)row * 256 + (c - 256)];
  else              v = ea[(size_t)row * 256 + (c - 512)];
  out[(size_t)row * 768 + c] = v;
}

// atom layer concat: [k[dst[e]] | k[src[e]] | ea[e]] -> bf16 [E,768]
__global__ void k_cat768_atom(const __bf16* __restrict__ kn,
                              const __bf16* __restrict__ ea,
                              const int* __restrict__ src,
                              const int* __restrict__ dst,
                              __bf16* __restrict__ out) {
  int e = blockIdx.x, c = threadIdx.x;  // blockDim = 768
  __bf16 v;
  if (c < 256)      v = kn[(size_t)dst[e] * 256 + c];
  else if (c < 512) v = kn[(size_t)src[e] * 256 + (c - 256)];
  else              v = ea[(size_t)e * 256 + (c - 512)];
  out[(size_t)e * 768 + c] = v;
}

// block(256)-wide mean/var over one 256-wide row (wave32 shfl reduction)
__device__ __forceinline__ void blockMeanVar(float v, float* s2, float& mean,
                                             float& var) {
  float s = v, q = v * v;
#pragma unroll
  for (int o = 16; o > 0; o >>= 1) {
    s += __shfl_xor(s, o, 32);
    q += __shfl_xor(q, o, 32);
  }
  int lane = threadIdx.x & 31, w = threadIdx.x >> 5;
  if (lane == 0) { s2[w] = s; s2[8 + w] = q; }
  __syncthreads();
  if (threadIdx.x < 32) {
    float ss = (lane < 8) ? s2[lane] : 0.f;
    float qq = (lane < 8) ? s2[8 + lane] : 0.f;
#pragma unroll
    for (int o = 4; o > 0; o >>= 1) {
      ss += __shfl_xor(ss, o, 32);
      qq += __shfl_xor(qq, o, 32);
    }
    if (lane == 0) { s2[0] = ss; s2[1] = qq; }
  }
  __syncthreads();
  mean = s2[0] * (1.f / 256.f);
  var  = s2[1] * (1.f / 256.f) - mean * mean;
}

// msg *= sigmoid(LN(q*key*scale)) -> bf16.  mode 0: qrow=row%E, 1: qrow=dst[row]
__global__ void k_gate(const float* __restrict__ q, const float* __restrict__ key,
                       const float* __restrict__ msg, const float* __restrict__ g,
                       const float* __restrict__ b, const int* __restrict__ dst,
                       int E, int mode, __bf16* __restrict__ out) {
  __shared__ float s2[16];
  int row = blockIdx.x, c = threadIdx.x;
  int qrow = (mode == 0) ? (row % E) : dst[row];
  float a = q[(size_t)qrow * 256 + c] * key[(size_t)row * 256 + c] * 0.0625f;
  float m, vv;
  blockMeanVar(a, s2, m, vv);
  float ln = (a - m) * rsqrtf(vv + 1e-5f) * g[c] + b[c];
  float gate = 1.f / (1.f + __expf(-ln));
  out[(size_t)row * 256 + c] = f2bf(msg[(size_t)row * 256 + c] * gate);
}

__global__ void k_lnrows(const float* __restrict__ x, const float* __restrict__ g,
                         const float* __restrict__ b, float* __restrict__ out) {
  __shared__ float s2[16];
  int row = blockIdx.x, c = threadIdx.x;
  float v = x[(size_t)row * 256 + c];
  float m, vv;
  blockMeanVar(v, s2, m, vv);
  out[(size_t)row * 256 + c] = (v - m) * rsqrtf(vv + 1e-5f) * g[c] + b[c];
}

// i-major sum over the 3 neighbor blocks
__global__ void k_sum3(const float* __restrict__ x, float* __restrict__ oF,
                       __bf16* __restrict__ oH, int E) {
  int i = blockIdx.x * 256 + threadIdx.x;
  if (i >= E * 256) return;
  float s = x[i] + x[(size_t)E * 256 + i] + x[(size_t)2 * E * 256 + i];
  oF[i] = s;
  oH[i] = f2bf(s);
}

__global__ void k_scatter_add(const float* __restrict__ x,
                              const int* __restrict__ dst,
                              float* __restrict__ agg, int E) {
  int i = blockIdx.x * 256 + threadIdx.x;
  if (i >= E * 256) return;
  int e = i >> 8, c = i & 255;
  atomicAdd(&agg[(size_t)dst[e] * 256 + c], x[i]);
}

__global__ void k_bnstats(const float* __restrict__ x, int M,
                          float* __restrict__ cs, float* __restrict__ cq) {
  int c = threadIdx.x;
  int r0 = blockIdx.x * 256;
  int r1 = r0 + 256; if (r1 > M) r1 = M;
  float s = 0.f, q = 0.f;
  for (int r = r0; r < r1; ++r) {
    float v = x[(size_t)r * 256 + c];
    s += v; q += v * v;
  }
  atomicAdd(&cs[c], s);
  atomicAdd(&cq[c], q);
}

// out = softplus(res + BN(x)); writes f32 + bf16 (res may alias oF)
__global__ void k_bnapply(const float* __restrict__ x, const float* __restrict__ res,
                          const float* __restrict__ cs, const float* __restrict__ cq,
                          const float* __restrict__ g, const float* __restrict__ b,
                          int M, float* __restrict__ oF, __bf16* __restrict__ oH) {
  int i = blockIdx.x * 256 + threadIdx.x;
  if (i >= M * 256) return;
  int c = i & 255;
  float inv = 1.f / (float)M;
  float m = cs[c] * inv;
  float v = cq[c] * inv - m * m;
  float y = (x[i] - m) * rsqrtf(v + 1e-5f) * g[c] + b[c];
  float o = softplusf(res[i] + y);
  oF[i] = o;
  oH[i] = f2bf(o);
}

__global__ void k_pool_acc(const float* __restrict__ node,
                           const int* __restrict__ batch,
                           float* __restrict__ sums, float* __restrict__ cnt,
                           int N) {
  int i = blockIdx.x * 256 + threadIdx.x;
  if (i >= N * 256) return;
  int v = i >> 8, c = i & 255;
  int g = batch[v];
  atomicAdd(&sums[(size_t)g * 256 + c], node[i]);
  if (c == 0) atomicAdd(&cnt[g], 1.f);
}

__global__ void k_pool_fin(const float* __restrict__ sums,
                           const float* __restrict__ cnt,
                           float* __restrict__ out) {
  int i = blockIdx.x * 256 + threadIdx.x;
  if (i >= NGRAPH * 256) return;
  out[i] = sums[i] / fmaxf(cnt[i >> 8], 1.f);
}

// =====================================================================
// Host orchestration
// =====================================================================
extern "C" void kernel_launch(void* const* d_in, const int* in_sizes, int n_in,
                              void* d_out, int out_size, void* d_ws,
                              size_t ws_size, hipStream_t stream) {
  (void)in_sizes; (void)n_in; (void)out_size; (void)ws_size;
  const int N = NNODES, E = NEDGES, G = NGRAPH;
  const int T = 3 * E;

  const float* x     = (const float*)d_in[0];
  const int*   eidx  = (const int*)d_in[1];
  const float* eattr = (const float*)d_in[2];
  const float* nlen  = (const float*)d_in[3];
  const float* nang  = (const float*)d_in[4];
  const int*   batch = (const int*)d_in[5];
  const int* src = eidx;
  const int* dst = eidx + E;

  // ---- walk params (insertion order, depth-first; None bias skipped) ----
  int pi = 6;
  auto P = [&]() -> const float* { return (const float*)d_in[pi++]; };
  struct LinP { const float* w; const float* b; };
  auto PL = [&]() { LinP l; l.w = P(); l.b = P(); return l; };

  const float* w_atom = P(); const float* b_atom = P();
  const float* w_eemb = P(); const float* b_eemb = P();
  const float* w_aemb = P(); const float* b_aemb = P();

  struct ELp {
    const float* lemb; LinP elen, q, k, v, ke[3], ve[3];
    const float* edge_w;
    LinP ku0, ku1, mu0, mu1, mlin;
    const float *mlng, *mlnb, *attg, *attb; LinP cat; const float *bng, *bnb;
  } el[2];
  struct ALp {
    LinP q, k, v, edge, ku0, ku1, mu0, mu1, mlin;
    const float *mlng, *mlnb, *attg, *attb; LinP cat; const float *bng, *bnb;
  } al[2];

  for (int l = 0; l < 2; ++l) {
    el[l].lemb = P();
    el[l].elen = PL(); el[l].q = PL(); el[l].k = PL(); el[l].v = PL();
    for (int i = 0; i < 3; ++i) el[l].ke[i] = PL();
    for (int i = 0; i < 3; ++i) el[l].ve[i] = PL();
    el[l].edge_w = P();                 // lin_edge: bias=None
    el[l].ku0 = PL(); el[l].ku1 = PL();
    el[l].mu0 = PL(); el[l].mu1 = PL();
    el[l].mlin = PL();
    el[l].mlng = P(); el[l].mlnb = P();
    el[l].attg = P(); el[l].attb = P();
    el[l].cat = PL();
    el[l].bng = P(); el[l].bnb = P();
  }
  for (int l = 0; l < 2; ++l) {
    al[l].q = PL(); al[l].k = PL(); al[l].v = PL(); al[l].edge = PL();
    al[l].ku0 = PL(); al[l].ku1 = PL(); al[l].mu0 = PL(); al[l].mu1 = PL();
    al[l].mlin = PL();
    al[l].mlng = P(); al[l].mlnb = P(); al[l].attg = P(); al[l].attb = P();
    al[l].cat = PL(); al[l].bng = P(); al[l].bnb = P();
  }

  // ---- scratch bump allocator ----
  char* bws = (char*)d_ws;
  size_t off = 0;
  auto alloc = [&](size_t bytes) -> void* {
    size_t o = (off + 255) & ~(size_t)255;
    off = o + bytes;
    return bws + o;
  };

  auto prep = [&](const float* w, int K) -> __bf16* {
    int Kp = (K + 31) & ~31;
    __bf16* wt = (__bf16*)alloc((size_t)Kp * 256 * 2);
    k_wprep<<<dim3(Kp), dim3(256), 0, stream>>>(w, wt, K, Kp);
    return wt;
  };

  __bf16* wt_atom = prep(w_atom, 92);
  __bf16* wt_eemb = prep(w_eemb, 256);
  __bf16* wt_aemb = prep(w_aemb, 256);

  struct ELw { __bf16 *elen, *q, *k, *v, *ke[3], *ve[3], *edge, *ku0, *ku1, *mu0, *mu1, *mlin, *cat; } elw[2];
  struct ALw { __bf16 *q, *k, *v, *edge, *ku0, *ku1, *mu0, *mu1, *mlin, *cat; } alw[2];
  for (int l = 0; l < 2; ++l) {
    elw[l].elen = prep(el[l].elen.w, 288);
    elw[l].q = prep(el[l].q.w, 256);
    elw[l].k = prep(el[l].k.w, 256);
    elw[l].v = prep(el[l].v.w, 256);
    for (int i = 0; i < 3; ++i) elw[l].ke[i] = prep(el[l].ke[i].w, 256);
    for (int i = 0; i < 3; ++i) elw[l].ve[i] = prep(el[l].ve[i].w, 256);
    elw[l].edge = prep(el[l].edge_w, 256);
    elw[l].ku0 = prep(el[l].ku0.w, 768); elw[l].ku1 = prep(el[l].ku1.w, 256);
    elw[l].mu0 = prep(el[l].mu0.w, 768); elw[l].mu1 = prep(el[l].mu1.w, 256);
    elw[l].mlin = prep(el[l].mlin.w, 256);
    elw[l].cat = prep(el[l].cat.w, 256);
  }
  for (int l = 0; l < 2; ++l) {
    alw[l].q = prep(al[l].q.w, 256);
    alw[l].k = prep(al[l].k.w, 256);
    alw[l].v = prep(al[l].v.w, 256);
    alw[l].edge = prep(al[l].edge.w, 256);
    alw[l].ku0 = prep(al[l].ku0.w, 768); alw[l].ku1 = prep(al[l].ku1.w, 256);
    alw[l].mu0 = prep(al[l].mu0.w, 768); alw[l].mu1 = prep(al[l].mu1.w, 256);
    alw[l].mlin = prep(al[l].mlin.w, 256);
    alw[l].cat = prep(al[l].cat.w, 256);
  }

  // ---- activation buffers ----
  float*  nodeF   = (float*)alloc((size_t)N * 256 * 4);
  __bf16* nodeB   = (__bf16*)alloc((size_t)N * 256 * 2);
  float*  edgeF   = (float*)alloc((size_t)E * 256 * 4);
  __bf16* edgeB   = (__bf16*)alloc((size_t)E * 256 * 2);
  __bf16* neilenB = (__bf16*)alloc((size_t)T * 256 * 2);
  __bf16* neiangB = (__bf16*)alloc((size_t)T * 256 * 2);
  __bf16* xpack   = (__bf16*)alloc((size_t)N * 96 * 2);
  float*  QF      = (float*)alloc((size_t)E * 256 * 4);
  __bf16* KXB     = (__bf16*)alloc((size_t)E * 256 * 2);
  __bf16* VXB     = (__bf16*)alloc((size_t)E * 256 * 2);
  __bf16* ELB     = (__bf16*)alloc((size_t)T * 256 * 2);
  __bf16* KYB     = (__bf16*)alloc((size_t)T * 256 * 2);
  __bf16* VYB     = (__bf16*)alloc((size_t)T * 256 * 2);
  __bf16* EAB     = (__bf16*)alloc((size_t)T * 256 * 2);
  __bf16* HIDB    = (__bf16*)alloc((size_t)T * 256 * 2);
  __bf16* GMB     = (__bf16*)alloc((size_t)T * 256 * 2);
  __bf16* CAT288  = (__bf16*)alloc((size_t)T * 288 * 2);
  __bf16* CAT768  = (__bf16*)alloc((size_t)T * 768 * 2);
  __bf16* RBFB    = CAT768;              // reuse: RBF phase precedes CAT768 use
  float*  KEYF    = (float*)alloc((size_t)T * 256 * 4);
  float*  MSGF    = (float*)alloc((size_t)T * 256 * 4);
  float*  M2F     = KEYF;                // reuse (KEYF dead after gate)
  float*  LNF     = MSGF;                // reuse (MSGF dead after gate)
  float*  SUMF    = (float*)alloc((size_t)E * 256 * 4);
  __bf16* SUMB    = (__bf16*)alloc((size_t)E * 256 * 2);
  float*  OCF     = (float*)alloc((size_t)E * 256 * 4);
  float*  AGGF    = (float*)alloc((size_t)N * 256 * 4);
  float*  colS    = (float*)alloc(512 * 4);
  float*  colQ    = colS + 256;
  float*  poolS   = (float*)alloc(((size_t)G * 256 + G) * 4);
  float*  poolC   = poolS + (size_t)G * 256;

  auto gemm = [&](const __bf16* A, __bf16* Wt, const float* bias, int M, int K,
                  int epi, float* oF, __bf16* oH) {
    int Kp = (K + 31) & ~31;
    dim3 grid((M + 127) / 128);
    if (epi == 1)      k_gemm<1><<<grid, 256, 0, stream>>>(A, Wt, bias, oF, oH, M, Kp);
    else if (epi == 2) k_gemm<2><<<grid, 256, 0, stream>>>(A, Wt, bias, oF, oH, M, Kp);
    else               k_gemm<0><<<grid, 256, 0, stream>>>(A, Wt, bias, oF, oH, M, Kp);
  };
  auto blocks = [](size_t n) { return dim3((unsigned)((n + 255) / 256)); };

  // ---- stage 1: embeddings ----
  k_pack_x<<<blocks((size_t)N * 96), 256, 0, stream>>>(x, xpack, N);
  gemm(xpack, wt_atom, b_atom, N, 96, 0, nodeF, nodeB);

  k_rbf<<<dim3(E), 256, 0, stream>>>(eattr, RBFB, E, 0, -4.f, 4.f);
  gemm(RBFB, wt_eemb, b_eemb, E, 256, 2, edgeF, edgeB);

  k_rbf<<<dim3(T), 256, 0, stream>>>(nlen, RBFB, E, 1, -4.f, 4.f);
  gemm(RBFB, wt_eemb, b_eemb, T, 256, 2, nullptr, neilenB);

  k_rbf<<<dim3(T), 256, 0, stream>>>(nang, RBFB, E, 1, -1.f, 1.f);
  gemm(RBFB, wt_aemb, b_aemb, T, 256, 2, nullptr, neiangB);

  // ---- layers ----
  for (int l = 0; l < 2; ++l) {
    // ====== UpdateEdge ======
    {
      const ELp& p = el[l]; const ELw& w = elw[l];
      k_cat288<<<dim3(T), 288, 0, stream>>>(neilenB, p.lemb, CAT288, E);
      gemm(CAT288, w.elen, p.elen.b, T, 288, 0, nullptr, ELB);
      gemm(edgeB, w.q, p.q.b, E, 256, 0, QF, nullptr);
      gemm(edgeB, w.k, p.k.b, E, 256, 0, nullptr, KXB);
      gemm(edgeB, w.v, p.v.b, E, 256, 0, nullptr, VXB);
      for (int i = 0; i < 3; ++i) {
        gemm(ELB + (size_t)i * E * 256, w.ke[i], p.ke[i].b, E, 256, 0,
             nullptr, KYB + (size_t)i * E * 256);
        gemm(ELB + (size_t)i * E * 256, w.ve[i], p.ve[i].b, E, 256, 0,
             nullptr, VYB + (size_t)i * E * 256);
      }
      gemm(neiangB, w.edge, nullptr, T, 256, 0, nullptr, EAB);

      k_cat768_edge<<<dim3(T), 768, 0, stream>>>(KXB, KYB, EAB, CAT768, E);
      gemm(CAT768, w.ku0, p.ku0.b, T, 768, 1, nullptr, HIDB);
      gemm(HIDB, w.ku1, p.ku1.b, T, 256, 0, KEYF, nullptr);
      k_cat768_edge<<<dim3(T), 768, 0, stream>>>(VXB, VYB, EAB, CAT768, E);
      gemm(CAT768, w.mu0, p.mu0.b, T, 768, 1, nullptr, HIDB);
      gemm(HIDB, w.mu1, p.mu1.b, T, 256, 0, MSGF, nullptr);

      k_gate<<<dim3(T), 256, 0, stream>>>(QF, KEYF, MSGF, p.attg, p.attb,
                                          nullptr, E, 0, GMB);
      gemm(GMB, w.mlin, p.mlin.b, T, 256, 0, M2F, nullptr);
      k_lnrows<<<dim3(T), 256, 0, stream>>>(M2F, p.mlng, p.mlnb, LNF);
      k_sum3<<<blocks((size_t)E * 256), 256, 0, stream>>>(LNF, SUMF, SUMB, E);
      gemm(SUMB, w.cat, p.cat.b, E, 256, 0, OCF, nullptr);
      k_zero<<<dim3(2), 256, 0, stream>>>(colS, 512);
      k_bnstats<<<dim3((E + 255) / 256), 256, 0, stream>>>(OCF, E, colS, colQ);
      k_bnapply<<<blocks((size_t)E * 256), 256, 0, stream>>>(
          OCF, edgeF, colS, colQ, p.bng, p.bnb, E, edgeF, edgeB);
    }
    // ====== TransformerConv ======
    {
      const ALp& p = al[l]; const ALw& w = alw[l];
      gemm(nodeB, w.q, p.q.b, N, 256, 0, QF, nullptr);
      gemm(nodeB, w.k, p.k.b, N, 256, 0, nullptr, KXB);
      gemm(nodeB, w.v, p.v.b, N, 256, 0, nullptr, VXB);
      gemm(edgeB, w.edge, p.edge.b, E, 256, 0, nullptr, EAB);

      k_cat768_atom<<<dim3(E), 768, 0, stream>>>(KXB, EAB, src, dst, CAT768);
      gemm(CAT768, w.ku0, p.ku0.b, E, 768, 1, nullptr, HIDB);
      gemm(HIDB, w.ku1, p.ku1.b, E, 256, 0, KEYF, nullptr);
      k_cat768_atom<<<dim3(E), 768, 0, stream>>>(VXB, EAB, src, dst, CAT768);
      gemm(CAT768, w.mu0, p.mu0.b, E, 768, 1, nullptr, HIDB);
      gemm(HIDB, w.mu1, p.mu1.b, E, 256, 0, MSGF, nullptr);

      k_gate<<<dim3(E), 256, 0, stream>>>(QF, KEYF, MSGF, p.attg, p.attb,
                                          dst, E, 1, GMB);
      gemm(GMB, w.mlin, p.mlin.b, E, 256, 0, M2F, nullptr);
      k_lnrows<<<dim3(E), 256, 0, stream>>>(M2F, p.mlng, p.mlnb, LNF);
      k_zero<<<blocks((size_t)N * 256), 256, 0, stream>>>(AGGF, N * 256);
      k_scatter_add<<<blocks((size_t)E * 256), 256, 0, stream>>>(LNF, dst, AGGF, E);
      k_cvt<<<blocks((size_t)N * 256), 256, 0, stream>>>(AGGF, SUMB, N * 256);
      gemm(SUMB, w.cat, p.cat.b, N, 256, 0, OCF, nullptr);
      k_zero<<<dim3(2), 256, 0, stream>>>(colS, 512);
      k_bnstats<<<dim3((N + 255) / 256), 256, 0, stream>>>(OCF, N, colS, colQ);
      k_bnapply<<<blocks((size_t)N * 256), 256, 0, stream>>>(
          OCF, nodeF, colS, colQ, p.bng, p.bnb, N, nodeF, nodeB);
    }
  }

  // ---- mean pool per graph ----
  k_zero<<<blocks((size_t)G * 256 + G), 256, 0, stream>>>(poolS, G * 256 + G);
  k_pool_acc<<<blocks((size_t)N * 256), 256, 0, stream>>>(nodeF, batch, poolS, poolC, N);
  k_pool_fin<<<blocks((size_t)G * 256), 256, 0, stream>>>(poolS, poolC, (float*)d_out);
}
